// MSDeformAttnTransformerEncoderLayer_55542517072596
// MI455X (gfx1250) — matrix operand order repaired and checked by reading, not compiled, not run on hardware
//
#include <hip/hip_runtime.h>
#include <hip/hip_bf16.h>
#include <math.h>
#include <stdint.h>

// ---------------------------------------------------------------------------
// Problem constants (match the reference)
// ---------------------------------------------------------------------------
#define BB   2
#define LL   13294
#define DD   256
#define NH   8
#define NL   4
#define NP   4
#define HD   32
#define FFN  1024
#define EPS_LN 1e-5f

__constant__ int c_H[NL]  = {100, 50, 25, 13};
__constant__ int c_W[NL]  = {100, 50, 25, 13};
__constant__ int c_ST[NL] = {0, 10000, 12500, 13125};

// ---------------------------------------------------------------------------
// bf16 WMMA types / helpers
// ---------------------------------------------------------------------------
typedef __bf16 bf16_t;
typedef __attribute__((ext_vector_type(16))) __bf16 v16bf;
typedef __attribute__((ext_vector_type(8)))  __bf16 v8bf;
typedef __attribute__((ext_vector_type(8)))  float  v8f;

// Pack two f32 into two RNE-rounded bf16 in one dword (lo = a, hi = b).
static __device__ inline unsigned pack2bf(float a, float b) {
    unsigned ua = __builtin_bit_cast(unsigned, a);
    unsigned ub = __builtin_bit_cast(unsigned, b);
    ua = ua + 0x7FFFu + ((ua >> 16) & 1u);
    ub = ub + 0x7FFFu + ((ub >> 16) & 1u);
    return (ua >> 16) | (ub & 0xFFFF0000u);
}
static __device__ inline bf16_t f2bf(float a) {
    unsigned ua = __builtin_bit_cast(unsigned, a);
    ua = ua + 0x7FFFu + ((ua >> 16) & 1u);
    return __builtin_bit_cast(bf16_t, (unsigned short)(ua >> 16));
}

// CDNA5 async global->LDS copy of 16B (tracked by ASYNCcnt).
static __device__ inline void async_load16(unsigned lds_off, const void* gaddr) {
    asm volatile("global_load_async_to_lds_b128 %0, %1, off"
                 :: "v"(lds_off), "v"(gaddr) : "memory");
}
static __device__ inline void wait_async0() {
    asm volatile("s_wait_asynccnt 0x0" ::: "memory");
}

// ---------------------------------------------------------------------------
// Tiled GEMM:  C[M,N] = A[M,K](bf16) @ W[N,K](bf16)^T + bias, opt. ReLU
// bf16 WMMA compute, f32 accumulate; output f32 or bf16.
// Block: 256 threads = 8 waves (4x2) -> 64x64 C tile, TK=64:
//   each wave holds two 16x16 accumulators -> 4 WMMAs per barrier pair.
// Staging: async global->LDS b128 copies (no VGPR round-trip).
// Preconditions: K % 64 == 0, N % 64 == 0 (M may be ragged).
// ---------------------------------------------------------------------------
#define TM 64
#define TN 64
#define TK 64
#define LDP (TK + 8)   // 72 bf16 = 144B row pitch: 16B multiple, bank-spread

template<bool RELU, bool BF16_OUT>
__global__ __launch_bounds__(256)
void gemm_bf16_wmma(const bf16_t* __restrict__ A,   // M x K bf16
                    const bf16_t* __restrict__ W,   // N x K bf16
                    const float*  __restrict__ bias,
                    void* __restrict__ Cv,          // f32 or bf16 per BF16_OUT
                    int M, int N, int K)
{
    __shared__ __align__(16) bf16_t sA[TM][LDP];
    __shared__ __align__(16) bf16_t sB[TN][LDP];

    const int tid  = threadIdx.x;
    const int wave = tid >> 5;       // 0..7
    const int lane = tid & 31;
    const int wm   = wave >> 1;      // 0..3  M-subtile
    const int wn   = wave & 1;       // 0..1  first of two N-subtiles (wn, wn+2)
    const int bm   = blockIdx.x * TM;
    const int bn   = blockIdx.y * TN;

    // Staging decomposition: 64 rows x 8 chunks(16B) = 512 chunks per tile;
    // 256 threads -> 2 chunks each (rows r and r+32).
    const int r  = tid >> 3;         // 0..31
    const int ch = tid & 7;          // 0..7 (16B chunk = 8 bf16)

    v8f acc0 = {0.f,0.f,0.f,0.f,0.f,0.f,0.f,0.f};
    v8f acc1 = {0.f,0.f,0.f,0.f,0.f,0.f,0.f,0.f};

    for (int k0 = 0; k0 < K; k0 += TK) {
        // ---- stage A tile (async b128, zero-fill ragged rows) ----
        #pragma unroll
        for (int i = 0; i < 2; ++i) {
            const int rr = r + i * 32;
            const int gm = bm + rr;
            const unsigned loff = (unsigned)(uintptr_t)&sA[rr][ch * 8];
            if (gm < M) {
                async_load16(loff, A + (size_t)gm * K + k0 + ch * 8);
            } else {
                *reinterpret_cast<uint4*>(&sA[rr][ch * 8]) = make_uint4(0u,0u,0u,0u);
            }
        }
        // ---- stage B tile (async b128; N%64==0 so always in range) ----
        #pragma unroll
        for (int i = 0; i < 2; ++i) {
            const int rr = r + i * 32;
            const unsigned loff = (unsigned)(uintptr_t)&sB[rr][ch * 8];
            async_load16(loff, W + (size_t)(bn + rr) * K + k0 + ch * 8);
        }
        // prefetch next K-tile into GL2
        if (k0 + TK < K) {
            __builtin_prefetch(A + (size_t)(bm + r) * K + k0 + TK + ch * 8, 0, 1);
            __builtin_prefetch(W + (size_t)(bn + r) * K + k0 + TK + ch * 8, 0, 1);
        }
        wait_async0();
        __syncthreads();

        // ---- 2 K-steps x (1 A-frag -> 2 B-frags) = 4 WMMAs ----
        #pragma unroll
        for (int ks = 0; ks < 2; ++ks) {
            const int kc = ks * 32;
            v16bf afrag;
            {
                const int m  = lane & 15;
                const int kb = (lane >> 4) * 16;          // 0 or 16
                const bf16_t* row = &sA[wm * 16 + m][kc];
                v8bf lo = *reinterpret_cast<const v8bf*>(row + kb);
                v8bf hi = *reinterpret_cast<const v8bf*>(row + kb + 8);
                #pragma unroll
                for (int e = 0; e < 8; ++e) { afrag[e] = lo[e]; afrag[e+8] = hi[e]; }
            }
            const int n   = lane & 15;
            const int kb2 = (lane >> 4) * 16;             // 0 or 16
            {
                const bf16_t* row = &sB[wn * 16 + n][kc];
                v16bf bfrag;
                v8bf lo = *reinterpret_cast<const v8bf*>(row + kb2);
                v8bf hi = *reinterpret_cast<const v8bf*>(row + kb2 + 8);
                #pragma unroll
                for (int e = 0; e < 8; ++e) { bfrag[e] = lo[e]; bfrag[e+8] = hi[e]; }
                acc0 = __builtin_amdgcn_wmma_f32_16x16x32_bf16(
                           false, afrag, false, bfrag, (short)0, acc0, false, false);
            }
            {
                const bf16_t* row = &sB[(wn + 2) * 16 + n][kc];
                v16bf bfrag;
                v8bf lo = *reinterpret_cast<const v8bf*>(row + kb2);
                v8bf hi = *reinterpret_cast<const v8bf*>(row + kb2 + 8);
                #pragma unroll
                for (int e = 0; e < 8; ++e) { bfrag[e] = lo[e]; bfrag[e+8] = hi[e]; }
                acc1 = __builtin_amdgcn_wmma_f32_16x16x32_bf16(
                           false, afrag, false, bfrag, (short)0, acc1, false, false);
            }
        }
        __syncthreads();
    }

    // ---- epilogue: C/D layout: vgpr v -> M = v + 8*(lane>=16), N = lane&15 ----
    const int moff = (lane >> 4) * 8;
    const int cmb  = bm + wm * 16 + moff;
    #pragma unroll
    for (int j = 0; j < 2; ++j) {
        const v8f& acc = j ? acc1 : acc0;
        const int cn = bn + (wn + 2 * j) * 16 + (lane & 15);
        const float bv = bias[cn];
        #pragma unroll
        for (int v = 0; v < 8; ++v) {
            const int cm = cmb + v;
            if (cm < M) {
                float rr = acc[v] + bv;
                if (RELU) rr = fmaxf(rr, 0.f);
                if (BF16_OUT) {
                    ((bf16_t*)Cv)[(size_t)cm * N + cn] = f2bf(rr);
                } else {
                    ((float*)Cv)[(size_t)cm * N + cn] = rr;
                }
            }
        }
    }
}

// ---------------------------------------------------------------------------
// f32 -> packed bf16 conversion, 8 elems / thread (b128 in, b128 out).
// ---------------------------------------------------------------------------
template<bool HAS_ADD>
__global__ __launch_bounds__(256)
void cvt_bf16_kernel(const float* __restrict__ a, const float* __restrict__ b,
                     uint4* __restrict__ out, int n8)
{
    int i = blockIdx.x * blockDim.x + threadIdx.x;
    if (i >= n8) return;
    const float4* pa = reinterpret_cast<const float4*>(a) + (size_t)i * 2;
    float4 v0 = pa[0], v1 = pa[1];
    if (HAS_ADD) {
        const float4* pb = reinterpret_cast<const float4*>(b) + (size_t)i * 2;
        float4 w0 = pb[0], w1 = pb[1];
        v0.x += w0.x; v0.y += w0.y; v0.z += w0.z; v0.w += w0.w;
        v1.x += w1.x; v1.y += w1.y; v1.z += w1.z; v1.w += w1.w;
    }
    uint4 o;
    o.x = pack2bf(v0.x, v0.y); o.y = pack2bf(v0.z, v0.w);
    o.z = pack2bf(v1.x, v1.y); o.w = pack2bf(v1.z, v1.w);
    out[i] = o;
}

// Fused: src -> src_bf and (src+pos) -> q_bf in one pass over inputs.
__global__ __launch_bounds__(256)
void cvt_src_q_kernel(const float* __restrict__ src, const float* __restrict__ pos,
                      uint4* __restrict__ sbf, uint4* __restrict__ qbf, int n8)
{
    int i = blockIdx.x * blockDim.x + threadIdx.x;
    if (i >= n8) return;
    const float4* ps = reinterpret_cast<const float4*>(src) + (size_t)i * 2;
    const float4* pp = reinterpret_cast<const float4*>(pos) + (size_t)i * 2;
    float4 s0 = ps[0], s1 = ps[1];
    float4 p0 = pp[0], p1 = pp[1];
    uint4 os;
    os.x = pack2bf(s0.x, s0.y); os.y = pack2bf(s0.z, s0.w);
    os.z = pack2bf(s1.x, s1.y); os.w = pack2bf(s1.z, s1.w);
    sbf[i] = os;
    uint4 oq;
    oq.x = pack2bf(s0.x + p0.x, s0.y + p0.y); oq.y = pack2bf(s0.z + p0.z, s0.w + p0.w);
    oq.z = pack2bf(s1.x + p1.x, s1.y + p1.y); oq.w = pack2bf(s1.z + p1.z, s1.w + p1.w);
    qbf[i] = oq;
}

// ---------------------------------------------------------------------------
// Softmax over contiguous groups of 16 (per b,l,h). One thread per group.
// ---------------------------------------------------------------------------
__global__ __launch_bounds__(256)
void softmax16_kernel(float* __restrict__ p, int rows)
{
    int r = blockIdx.x * blockDim.x + threadIdx.x;
    if (r >= rows) return;
    float* q = p + (size_t)r * 16;
    float v[16], mx = -3.0e38f;
    #pragma unroll
    for (int i = 0; i < 16; ++i) { v[i] = q[i]; mx = fmaxf(mx, v[i]); }
    float s = 0.f;
    #pragma unroll
    for (int i = 0; i < 16; ++i) { v[i] = __expf(v[i] - mx); s += v[i]; }
    float inv = 1.f / s;
    #pragma unroll
    for (int i = 0; i < 16; ++i) q[i] = v[i] * inv;
}

// ---------------------------------------------------------------------------
// Deformable sampling: one wave32 per (b,l,h), lane = channel d (HD=32).
// value laid out (B, L, NH, HD) so gathers are 128B coalesced (L2-resident).
// ---------------------------------------------------------------------------
__global__ __launch_bounds__(256)
void msdeform_sample_kernel(const float* __restrict__ value,  // (B*L, NH*HD)
                            const float* __restrict__ off,    // (B*L, NH*NL*NP*2)
                            const float* __restrict__ aw,     // (B*L, NH*NL*NP)
                            const float* __restrict__ refp,   // (B, L, NL, 2)
                            float* __restrict__ out)          // (B*L, NH*HD)
{
    const int wid  = blockIdx.x * (blockDim.x >> 5) + (threadIdx.x >> 5);
    const int lane = threadIdx.x & 31;
    const int total = BB * LL * NH;
    if (wid >= total) return;

    const int h  = wid % NH;
    const int bl = wid / NH;          // b*L + l
    const int b  = bl / LL;

    const float* offr = off + (size_t)bl * (NH * NL * NP * 2) + h * (NL * NP * 2);
    const float* awr  = aw  + (size_t)bl * (NH * NL * NP)     + h * (NL * NP);
    const float* rpr  = refp + (size_t)bl * (NL * 2);

    float accum = 0.f;
    #pragma unroll
    for (int lvl = 0; lvl < NL; ++lvl) {
        const int Hh = c_H[lvl], Ww = c_W[lvl], st = c_ST[lvl];
        const float rx = rpr[lvl * 2 + 0];
        const float ry = rpr[lvl * 2 + 1];
        const float invW = 1.f / (float)Ww;
        const float invH = 1.f / (float)Hh;
        const float* vbase = value + ((size_t)(b * LL + st) * DD) + h * HD + lane;

        #pragma unroll
        for (int p = 0; p < NP; ++p) {
            const float ox = offr[(lvl * NP + p) * 2 + 0];
            const float oy = offr[(lvl * NP + p) * 2 + 1];
            const float w  = awr[lvl * NP + p];

            const float x = (rx + ox * invW) * (float)Ww - 0.5f;
            const float y = (ry + oy * invH) * (float)Hh - 0.5f;
            const float x0f = floorf(x), y0f = floorf(y);
            const float lx = x - x0f, ly = y - y0f;
            const int x0 = (int)x0f, y0 = (int)y0f;

            float acc_p = 0.f;
            #pragma unroll
            for (int t = 0; t < 4; ++t) {
                const int xi = x0 + (t & 1);
                const int yi = y0 + (t >> 1);
                const float wx = (t & 1) ? lx : (1.f - lx);
                const float wy = (t >> 1) ? ly : (1.f - ly);
                if (xi >= 0 && xi < Ww && yi >= 0 && yi < Hh) {
                    acc_p += wx * wy * vbase[(size_t)(yi * Ww + xi) * DD];
                }
            }
            accum += w * acc_p;
        }
    }
    out[(size_t)bl * DD + h * HD + lane] = accum;
}

// ---------------------------------------------------------------------------
// out = LayerNorm(a + b) * g + be.  One wave32 per 256-wide row.
// ---------------------------------------------------------------------------
__global__ __launch_bounds__(256)
void add_layernorm_kernel(const float* __restrict__ a,
                          const float* __restrict__ b,
                          const float* __restrict__ g,
                          const float* __restrict__ be,
                          float* __restrict__ out, int rows)
{
    const int wid  = blockIdx.x * (blockDim.x >> 5) + (threadIdx.x >> 5);
    const int lane = threadIdx.x & 31;
    if (wid >= rows) return;

    const size_t base = (size_t)wid * DD;
    float v[8];
    float s = 0.f;
    #pragma unroll
    for (int i = 0; i < 8; ++i) {
        int d = lane * 8 + i;
        v[i] = a[base + d] + b[base + d];
        s += v[i];
    }
    #pragma unroll
    for (int m = 16; m >= 1; m >>= 1) s += __shfl_xor(s, m, 32);
    const float mean = s * (1.f / (float)DD);

    float vs = 0.f;
    #pragma unroll
    for (int i = 0; i < 8; ++i) { float d0 = v[i] - mean; vs += d0 * d0; }
    #pragma unroll
    for (int m = 16; m >= 1; m >>= 1) vs += __shfl_xor(vs, m, 32);
    const float rstd = rsqrtf(vs * (1.f / (float)DD) + EPS_LN);

    #pragma unroll
    for (int i = 0; i < 8; ++i) {
        int d = lane * 8 + i;
        out[base + d] = (v[i] - mean) * rstd * g[d] + be[d];
    }
}

// ---------------------------------------------------------------------------
// Host-side orchestration
// ---------------------------------------------------------------------------
extern "C" void kernel_launch(void* const* d_in, const int* in_sizes, int n_in,
                              void* d_out, int out_size, void* d_ws, size_t ws_size,
                              hipStream_t stream)
{
    (void)in_sizes; (void)n_in; (void)out_size; (void)ws_size;

    const float* src    = (const float*)d_in[0];
    const float* pos    = (const float*)d_in[1];
    const float* refp   = (const float*)d_in[2];
    // d_in[3] spatial_shapes / d_in[4] level_start_index: compile-time constants
    const float* W_off  = (const float*)d_in[5];
    const float* b_off  = (const float*)d_in[6];
    const float* W_attn = (const float*)d_in[7];
    const float* b_attn = (const float*)d_in[8];
    const float* W_v    = (const float*)d_in[9];
    const float* b_v    = (const float*)d_in[10];
    const float* W_o    = (const float*)d_in[11];
    const float* b_o    = (const float*)d_in[12];
    const float* g1     = (const float*)d_in[13];
    const float* be1    = (const float*)d_in[14];
    const float* W_fc1  = (const float*)d_in[15];
    const float* b_fc1  = (const float*)d_in[16];
    const float* W_fc2  = (const float*)d_in[17];
    const float* b_fc2  = (const float*)d_in[18];
    const float* g2     = (const float*)d_in[19];
    const float* be2    = (const float*)d_in[20];

    const int M = BB * LL;                 // 26588 rows
    char* ws = (char*)d_ws;

    // Workspace layout (bytes, 256B-aligned blocks):
    size_t off_b = 0;
    auto alloc = [&](size_t bytes) { size_t o = off_b; off_b += (bytes + 255) & ~(size_t)255; return o; };
    const size_t o_value  = alloc((size_t)M * DD * 4);    // f32 value
    const size_t o_off    = alloc((size_t)M * DD * 4);    // f32 off (later attn_out / y)
    const size_t o_aw     = alloc((size_t)M * 128 * 4);   // f32 logits -> softmax
    const size_t o_samp   = alloc((size_t)M * DD * 4);    // f32 sampled
    const size_t o_x      = alloc((size_t)M * DD * 4);    // f32 x
    const size_t o_srcbf  = alloc((size_t)M * DD * 2);    // bf16 src
    const size_t o_qbf    = alloc((size_t)M * DD * 2);    // bf16 q = src+pos
    const size_t o_sampbf = alloc((size_t)M * DD * 2);    // bf16 sampled
    const size_t o_xbf    = alloc((size_t)M * DD * 2);    // bf16 x
    const size_t o_hbf    = alloc((size_t)M * FFN * 2);   // bf16 h (fc1 out)
    const size_t o_wv     = alloc((size_t)DD * DD * 2);
    const size_t o_woff   = alloc((size_t)DD * DD * 2);
    const size_t o_wattn  = alloc((size_t)128 * DD * 2);
    const size_t o_wo     = alloc((size_t)DD * DD * 2);
    const size_t o_wfc1   = alloc((size_t)FFN * DD * 2);
    const size_t o_wfc2   = alloc((size_t)DD * FFN * 2);

    float*  f_value = (float*)(ws + o_value);
    float*  f_off   = (float*)(ws + o_off);
    float*  f_aw    = (float*)(ws + o_aw);
    float*  f_samp  = (float*)(ws + o_samp);
    float*  f_x     = (float*)(ws + o_x);
    bf16_t* b_src   = (bf16_t*)(ws + o_srcbf);
    bf16_t* b_q     = (bf16_t*)(ws + o_qbf);
    bf16_t* b_samp  = (bf16_t*)(ws + o_sampbf);
    bf16_t* b_x     = (bf16_t*)(ws + o_xbf);
    bf16_t* b_h     = (bf16_t*)(ws + o_hbf);
    bf16_t* b_wv    = (bf16_t*)(ws + o_wv);
    bf16_t* b_woff  = (bf16_t*)(ws + o_woff);
    bf16_t* b_wattn = (bf16_t*)(ws + o_wattn);
    bf16_t* b_wo    = (bf16_t*)(ws + o_wo);
    bf16_t* b_wfc1  = (bf16_t*)(ws + o_wfc1);
    bf16_t* b_wfc2  = (bf16_t*)(ws + o_wfc2);

    const int mt = (M + TM - 1) / TM;                      // 416 M-tiles
    const dim3 blk(256);
    const int nMD8 = M * DD / 8;                           // 8-elem groups

    // 0) conversions: activations + weights -> bf16
    cvt_src_q_kernel<<<(nMD8 + 255) / 256, blk, 0, stream>>>(
        src, pos, (uint4*)b_src, (uint4*)b_q, nMD8);
    cvt_bf16_kernel<false><<<(DD*DD/8 + 255) / 256, blk, 0, stream>>>(W_v,   nullptr, (uint4*)b_wv,   DD*DD/8);
    cvt_bf16_kernel<false><<<(DD*DD/8 + 255) / 256, blk, 0, stream>>>(W_off, nullptr, (uint4*)b_woff, DD*DD/8);
    cvt_bf16_kernel<false><<<(128*DD/8 + 255) / 256, blk, 0, stream>>>(W_attn, nullptr, (uint4*)b_wattn, 128*DD/8);
    cvt_bf16_kernel<false><<<(DD*DD/8 + 255) / 256, blk, 0, stream>>>(W_o,   nullptr, (uint4*)b_wo,   DD*DD/8);
    cvt_bf16_kernel<false><<<(FFN*DD/8 + 255) / 256, blk, 0, stream>>>(W_fc1, nullptr, (uint4*)b_wfc1, FFN*DD/8);
    cvt_bf16_kernel<false><<<(DD*FFN/8 + 255) / 256, blk, 0, stream>>>(W_fc2, nullptr, (uint4*)b_wfc2, DD*FFN/8);

    // 1) value = src @ W_v^T + b_v                     (M x 256)
    gemm_bf16_wmma<false, false><<<dim3(mt, DD / TN), blk, 0, stream>>>(
        b_src, b_wv, b_v, f_value, M, DD, DD);

    // 2) off = q @ W_off^T + b_off                    (M x 256)
    gemm_bf16_wmma<false, false><<<dim3(mt, DD / TN), blk, 0, stream>>>(
        b_q, b_woff, b_off, f_off, M, DD, DD);

    // 3) logits = q @ W_attn^T + b_attn               (M x 128)
    gemm_bf16_wmma<false, false><<<dim3(mt, 128 / TN), blk, 0, stream>>>(
        b_q, b_wattn, b_attn, f_aw, M, 128, DD);

    // 4) softmax over 16 per (b,l,h)
    {
        int rows = M * NH;
        softmax16_kernel<<<(rows + 255) / 256, blk, 0, stream>>>(f_aw, rows);
    }

    // 5) deformable bilinear sampling -> (B, L, NH*HD)
    {
        int waves = M * NH;                 // one wave per (b,l,h)
        int blocks = (waves + 7) / 8;       // 8 waves / 256-thread block
        msdeform_sample_kernel<<<blocks, blk, 0, stream>>>(
            f_value, f_off, f_aw, refp, f_samp);
    }

    // 6) attn_out = samp @ W_o^T + b_o  (reuse off buffer)
    cvt_bf16_kernel<false><<<(nMD8 + 255) / 256, blk, 0, stream>>>(
        f_samp, nullptr, (uint4*)b_samp, nMD8);
    gemm_bf16_wmma<false, false><<<dim3(mt, DD / TN), blk, 0, stream>>>(
        b_samp, b_wo, b_o, f_off, M, DD, DD);

    // 7) x = LN(src + attn_out)
    add_layernorm_kernel<<<(M + 7) / 8, blk, 0, stream>>>(
        src, f_off, g1, be1, f_x, M);
    cvt_bf16_kernel<false><<<(nMD8 + 255) / 256, blk, 0, stream>>>(
        f_x, nullptr, (uint4*)b_x, nMD8);

    // 8) h = relu(x @ W_fc1^T + b_fc1)                (M x 1024, bf16 out)
    gemm_bf16_wmma<true, true><<<dim3(mt, FFN / TN), blk, 0, stream>>>(
        b_x, b_wfc1, b_fc1, b_h, M, FFN, DD);

    // 9) y = h @ W_fc2^T + b_fc2  (reuse off buffer)
    gemm_bf16_wmma<false, false><<<dim3(mt, DD / TN), blk, 0, stream>>>(
        b_h, b_wfc2, b_fc2, f_off, M, DD, FFN);

    // 10) out = LN(x + y)
    add_layernorm_kernel<<<(M + 7) / 8, blk, 0, stream>>>(
        f_x, f_off, g2, be2, (float*)d_out, M);
}
